// GAE_8220567405314
// MI455X (gfx1250) — matrix operand order repaired and checked by reading, not compiled, not run on hardware
//
#include <hip/hip_runtime.h>
#include <hip/hip_bf16.h>

#define IN_CH 256
#define OUT_CH 64

typedef __attribute__((ext_vector_type(2)))  float    v2f;
typedef __attribute__((ext_vector_type(8)))  float    v8f;
typedef __attribute__((ext_vector_type(16))) __bf16   v16bf;
typedef __attribute__((ext_vector_type(4)))  unsigned v4u;
typedef __attribute__((ext_vector_type(8)))  int      v8i;
typedef __attribute__((ext_vector_type(4)))  int      v4i;

__device__ __forceinline__ float atomic_add_f32(float* p, float v) {
    return __hip_atomic_fetch_add(p, v, __ATOMIC_RELAXED, __HIP_MEMORY_SCOPE_AGENT);
}

// ---------------- degree / normalization ----------------
__global__ void k_degree(const long long* __restrict__ ei, float* __restrict__ deg, int E) {
    int e = blockIdx.x * blockDim.x + threadIdx.x;
    if (e < E) atomic_add_f32(&deg[(int)ei[E + e]], 1.0f);
}

__global__ void k_dinv(float* __restrict__ deg, int n) {
    int i = blockIdx.x * blockDim.x + threadIdx.x;
    if (i < n) deg[i] = rsqrtf(deg[i] + 1.0f);   // +1 = self loop; deg >= 1 always
}

// ---------------- f32 WMMA GEMM: Y[M,N] = X[M,K] @ W[K,N] ----------------
// block = 128 threads (4 waves); each wave owns one 16x16 output tile.
// grid = (M/16, N/64)
__global__ __launch_bounds__(128) void k_gemm_f32wmma(const float* __restrict__ X,
                                                      const float* __restrict__ W,
                                                      float* __restrict__ Y,
                                                      int K, int N) {
    const int tid  = threadIdx.x;
    const int wave = tid >> 5;
    const int lane = tid & 31;
    const int m    = lane & 15;
    const int hh   = lane >> 4;                  // lane half selects K pair / row+8
    const int i0   = blockIdx.x * 16;
    const int j    = blockIdx.y * 64 + wave * 16 + m;

    v8f acc = {};
    const float2* xrow = (const float2*)(X + (size_t)(i0 + m) * K);
    for (int k = 0; k < K; k += 4) {
        // A 16x4 f32: v0 = A[m][k+2*hh], v1 = A[m][k+2*hh+1]
        float2 av = xrow[(k >> 1) + hh];
        v2f a; a[0] = av.x; a[1] = av.y;
        // B 4x16 f32: v0 = B[k+2*hh][n], v1 = B[k+2*hh+1][n]
        v2f b;
        b[0] = W[(size_t)(k + 2 * hh)     * N + j];
        b[1] = W[(size_t)(k + 2 * hh + 1) * N + j];
        acc = __builtin_amdgcn_wmma_f32_16x16x4_f32(false, a, false, b, (short)0, acc,
                                                    false, false);
    }
#pragma unroll
    for (int r = 0; r < 8; ++r)
        Y[(size_t)(i0 + r + 8 * hh) * N + j] = acc[r];
}

// ---------------- edge scatter-add (256 channels: block per edge) ----------------
__global__ __launch_bounds__(256) void k_agg256(const long long* __restrict__ ei,
                                                const float* __restrict__ t,
                                                const float* __restrict__ dinv,
                                                float* __restrict__ agg, int E) {
    int e = blockIdx.x;
    int c = threadIdx.x;
    int s = (int)ei[e];
    int d = (int)ei[E + e];
    float norm = dinv[s] * dinv[d];
    atomic_add_f32(&agg[(size_t)d * 256 + c], t[(size_t)s * 256 + c] * norm);
}

// 64 channels: 4 edges per 256-thread block
__global__ __launch_bounds__(256) void k_agg64(const long long* __restrict__ ei,
                                               const float* __restrict__ t,
                                               const float* __restrict__ dinv,
                                               float* __restrict__ agg, int E) {
    int e = blockIdx.x * 4 + (threadIdx.x >> 6);
    int c = threadIdx.x & 63;
    int s = (int)ei[e];
    int d = (int)ei[E + e];
    float norm = dinv[s] * dinv[d];
    atomic_add_f32(&agg[(size_t)d * 64 + c], t[(size_t)s * 64 + c] * norm);
}

// ---------------- self-loop + bias + relu (layer 1, in place over agg) -----------
__global__ void k_combine_relu(const float* __restrict__ t1, const float* __restrict__ dinv,
                               const float* __restrict__ b, float* __restrict__ agg) {
    int idx = blockIdx.x * blockDim.x + threadIdx.x;
    int i = idx >> 8, c = idx & 255;
    float di = dinv[i];
    float v = agg[idx] + t1[idx] * di * di + b[c];
    agg[idx] = fmaxf(v, 0.0f);
}

// ---------------- self-loop + bias + bf16 hi/lo split (layer 2) ------------------
__global__ void k_combine_split(const float* __restrict__ t2, const float* __restrict__ dinv,
                                const float* __restrict__ b, const float* __restrict__ agg,
                                __bf16* __restrict__ zhi, __bf16* __restrict__ zlo) {
    int idx = blockIdx.x * blockDim.x + threadIdx.x;
    int i = idx >> 6, c = idx & 63;
    float di = dinv[i];
    float z = agg[idx] + t2[idx] * di * di + b[c];
    __bf16 h = (__bf16)z;
    zhi[idx] = h;
    zlo[idx] = (__bf16)(z - (float)h);         // residual for error compensation
}

// ---------------- TDM: load `rows` x 64 bf16 tile into padded LDS ----------------
// LDS destination rows are padded: 64 data halves + 8 pad halves (stride 72 = 144B),
// produced by the TDM pad feature (32 dwords data, then 4 dwords pad).
__device__ __forceinline__ void tdm_load_tile(unsigned lds_off, const void* gptr, int rows) {
    unsigned long long ga = (unsigned long long)(uintptr_t)gptr;
    v4u g0;
    g0[0] = 1u;                                        // count=1, user descriptor
    g0[1] = lds_off;                                   // lds_addr (bytes)
    g0[2] = (unsigned)(ga & 0xffffffffu);              // global_addr lo
    g0[3] = (unsigned)((ga >> 32) & 0x01ffffffu)       // global_addr hi (bits 56:32)
          | (2u << 30);                                // type = 2 ("image")
    v8i g1;
    g1[0] = (1 << 16)                                  // data_size = 2 bytes
          | (1 << 20)                                  // pad_enable
          | (4 << 22)                                  // pad_interval = 32 dwords
          | (3 << 25);                                 // pad_amount   = 4 dwords
    g1[1] = (64 << 16);                                // tensor_dim0 = 64 elements
    g1[2] = (int)(0x3000u << 16);                      // tensor_dim1 = 12288 (lo 16)
    g1[3] = (64 << 16);                                // tile_dim0 = 64 elements
    g1[4] = rows & 0xffff;                             // tile_dim1 = rows, tile_dim2 = 0
    g1[5] = 64;                                        // tensor_dim0_stride = 64
    g1[6] = (64 << 16);                                // tensor_dim1_stride = 64 (lo 16)
    g1[7] = 0;
    v4i z4 = {0, 0, 0, 0};
    v8i z8 = {0, 0, 0, 0, 0, 0, 0, 0};
    __builtin_amdgcn_tensor_load_to_lds(g0, g1, z4, z4, z8, 0);
}

__device__ __forceinline__ v16bf frag_b128x2(const __bf16* p0, const __bf16* p1) {
    union { v16bf v; uint4 q[2]; } u;                  // two ds_load_b128
    u.q[0] = *(const uint4*)p0;
    u.q[1] = *(const uint4*)p1;
    return u.v;
}

// ---------------- decode: sigmoid(z @ z^T), bf16 hi/lo compensated WMMA ----------
// block = 256 threads (8 waves). Block tile = 16 rows x 128 cols; wave w -> cols +16w.
#define LDSTR 72   // padded LDS row stride in halves (144 B = 36 banks, conflict-free)
__global__ __launch_bounds__(256) void k_decode(const __bf16* __restrict__ zhi,
                                                const __bf16* __restrict__ zlo,
                                                float* __restrict__ out, int n) {
    __shared__ __align__(16) __bf16 sAhi[16 * LDSTR],  sAlo[16 * LDSTR];
    __shared__ __align__(16) __bf16 sBhi[128 * LDSTR], sBlo[128 * LDSTR];
    __shared__ __align__(16) float  sOut[16 * 132];    // padded f32 staging for stores

    const int tid = threadIdx.x;
    const int i0  = blockIdx.x * 16;
    const int j0  = blockIdx.y * 128;

    if (tid < 32) {                                    // wave 0 drives the TDM
        tdm_load_tile((unsigned)(uintptr_t)sAhi, zhi + (size_t)i0 * 64, 16);
        tdm_load_tile((unsigned)(uintptr_t)sAlo, zlo + (size_t)i0 * 64, 16);
        tdm_load_tile((unsigned)(uintptr_t)sBhi, zhi + (size_t)j0 * 64, 128);
        tdm_load_tile((unsigned)(uintptr_t)sBlo, zlo + (size_t)j0 * 64, 128);
        __builtin_amdgcn_s_wait_tensorcnt(0);
    }
    __syncthreads();

    const int wave = tid >> 5;
    const int lane = tid & 31;
    const int nl   = lane & 15;
    const int hh   = lane >> 4;
    const int colL = wave * 16 + nl;

    v8f acc = {};
#pragma unroll
    for (int kb = 0; kb < 64; kb += 32) {
        // A 16x32 bf16: lane half hh -> halves [kb+8hh .. +7] and [kb+16+8hh .. +7]
        v16bf aH = frag_b128x2(&sAhi[nl * LDSTR + kb + hh * 8],
                               &sAhi[nl * LDSTR + kb + 16 + hh * 8]);
        v16bf aL = frag_b128x2(&sAlo[nl * LDSTR + kb + hh * 8],
                               &sAlo[nl * LDSTR + kb + 16 + hh * 8]);
        // B 32x16 bf16: lane half hh -> halves [kb+8hh .. +7] and [kb+16+8hh .. +7]
        v16bf bH = frag_b128x2(&sBhi[colL * LDSTR + kb + hh * 8],
                               &sBhi[colL * LDSTR + kb + 16 + hh * 8]);
        v16bf bL = frag_b128x2(&sBlo[colL * LDSTR + kb + hh * 8],
                               &sBlo[colL * LDSTR + kb + 16 + hh * 8]);
        acc = __builtin_amdgcn_wmma_f32_16x16x32_bf16(false, aH, false, bH, (short)0, acc, false, false);
        acc = __builtin_amdgcn_wmma_f32_16x16x32_bf16(false, aH, false, bL, (short)0, acc, false, false);
        acc = __builtin_amdgcn_wmma_f32_16x16x32_bf16(false, aL, false, bH, (short)0, acc, false, false);
    }

    // sigmoid into padded LDS staging (conflict-free: 132-dword row stride)
#pragma unroll
    for (int r = 0; r < 8; ++r) {
        float v = acc[r];
        sOut[(r + 8 * hh) * 132 + colL] = 1.0f / (1.0f + __expf(-v));
    }
    __syncthreads();

    // fully coalesced float4 stores: 16 rows x 512B contiguous per row
    for (int v = tid; v < 16 * 32; v += 256) {
        int row = v >> 5, q = v & 31;
        float4 val = *(const float4*)&sOut[row * 132 + q * 4];
        *(float4*)&out[(size_t)(i0 + row) * n + j0 + q * 4] = val;
    }
}

// =======================================================================
extern "C" void kernel_launch(void* const* d_in, const int* in_sizes, int n_in,
                              void* d_out, int out_size, void* d_ws, size_t ws_size,
                              hipStream_t stream) {
    const float*     x  = (const float*)d_in[0];
    const long long* ei = (const long long*)d_in[1];
    const float*     W1 = (const float*)d_in[2];
    const float*     b1 = (const float*)d_in[3];
    const float*     W2 = (const float*)d_in[4];
    const float*     b2 = (const float*)d_in[5];
    float*           out = (float*)d_out;

    const int n = in_sizes[0] / IN_CH;   // 12288
    const int E = in_sizes[1] / 2;       // 393216

    char* ws = (char*)d_ws;
    float*  deg  = (float*)ws;                         // n  (becomes dinv in place)
    float*  t1   = (float*)(ws + 64 * 1024);           // n*256
    float*  agg1 = t1 + (size_t)n * 256;               // n*256 (becomes h in place)
    float*  t2   = agg1 + (size_t)n * 256;             // n*64
    float*  agg2 = t2 + (size_t)n * 64;                // n*64 (becomes z pre-split)
    __bf16* zhi  = (__bf16*)(agg2 + (size_t)n * 64);   // n*64 bf16
    __bf16* zlo  = zhi + (size_t)n * 64;               // n*64 bf16

    (void)hipMemsetAsync(deg,  0, (size_t)n * sizeof(float), stream);
    (void)hipMemsetAsync(agg1, 0, (size_t)n * 256 * sizeof(float), stream);
    (void)hipMemsetAsync(agg2, 0, (size_t)n * 64 * sizeof(float), stream);

    k_degree<<<E / 256, 256, 0, stream>>>(ei, deg, E);
    k_dinv<<<n / 256, 256, 0, stream>>>(deg, n);

    // layer 1: t1 = x @ W1 (f32 WMMA)
    k_gemm_f32wmma<<<dim3(n / 16, 256 / 64), 128, 0, stream>>>(x, W1, t1, 256, 256);
    k_agg256<<<E, 256, 0, stream>>>(ei, t1, deg, agg1, E);
    k_combine_relu<<<(n * 256) / 256, 256, 0, stream>>>(t1, deg, b1, agg1);

    // layer 2: t2 = h @ W2 (f32 WMMA)
    k_gemm_f32wmma<<<dim3(n / 16, 1), 128, 0, stream>>>(agg1, W2, t2, 256, 64);
    k_agg64<<<E / 4, 256, 0, stream>>>(ei, t2, deg, agg2, E);
    k_combine_split<<<(n * 64) / 256, 256, 0, stream>>>(t2, deg, b2, agg2, zhi, zlo);

    // decode: sigmoid(z @ z^T), TDM-staged tiles, bf16 hi/lo compensated WMMA
    k_decode<<<dim3(n / 16, n / 128), 256, 0, stream>>>(zhi, zlo, out, n);
}
